// CausalAttention_78907139162308
// MI455X (gfx1250) — matrix operand order repaired and checked by reading, not compiled, not run on hardware
//
#include <hip/hip_runtime.h>
#include <hip/hip_bf16.h>

typedef __attribute__((ext_vector_type(16))) _Float16 v16h;
typedef __attribute__((ext_vector_type(8)))  _Float16 v8h;
typedef __attribute__((ext_vector_type(4)))  _Float16 v4h;
typedef __attribute__((ext_vector_type(8)))  float    v8f;
typedef __attribute__((ext_vector_type(4)))  float    v4f;
typedef __attribute__((ext_vector_type(4)))  unsigned int v4u;
typedef __attribute__((ext_vector_type(8)))  int      v8i;
typedef __attribute__((ext_vector_type(4)))  int      v4i;

#define B_DIM 4
#define T_DIM 2048
#define D_DIM 1024

// Padded K-tile row stride in LDS bytes: 2048 data + 2 pads of 16B (TDM pad:
// 4 dwords per 1024B interval). Keeps 16B alignment, spreads banks.
#define KROW_BYTES 2080
#define KBUF_HALFS (32 * KROW_BYTES / 2)   // 33280 halfs = 66560 B per buffer

__device__ __forceinline__ v8f wmma_f16(v16h a, v16h b, v8f c) {
  return __builtin_amdgcn_wmma_f32_16x16x32_f16(
      false, a, false, b, (short)0, c, false, false);
}

__device__ __forceinline__ unsigned lds_offset(const void* p) {
  // Flat shared pointer: LDS byte offset lives in addr[31:0] (ISA 10.2)
  return (unsigned)(unsigned long long)(uintptr_t)p;
}

// TDM 2D tile load: global (row-major, 2-byte elements) -> LDS, optional pad.
// D# layout per CDNA5 ISA sec.8 (groups 0/1; 2D -> rest 0).
__device__ __forceinline__ void tdm_load_2d_f16(
    unsigned lds_addr, const void* gptr,
    unsigned tile0, unsigned tile1,
    unsigned tensor0, unsigned tensor1, unsigned stride0_elems,
    int pad_en, unsigned pad_interval, unsigned pad_amount)
{
  unsigned long long ga = (unsigned long long)(uintptr_t)gptr;
  v4u g0;
  g0[0] = 1u;                                   // count=1, user flags 0
  g0[1] = lds_addr;                             // lds_addr [63:32]
  g0[2] = (unsigned)(ga & 0xffffffffu);         // global_addr lo
  g0[3] = (unsigned)((ga >> 32) & 0x01ffffffu)  // global_addr hi (57-bit)
          | (2u << 30);                         // type=2 ("image")
  unsigned d0 = (1u << 16);                     // data_size=1 -> 2 bytes
  if (pad_en) d0 |= (1u << 20) | (pad_interval << 22) | (pad_amount << 25);
  v8i g1;
  g1[0] = (int)d0;
  g1[1] = (int)((tensor0 & 0xffffu) << 16);                     // abar=0 | dim0.lo
  g1[2] = (int)((tensor0 >> 16) | ((tensor1 & 0xffffu) << 16)); // dim0.hi | dim1.lo
  g1[3] = (int)((tensor1 >> 16) | (tile0 << 16));               // dim1.hi | tile0
  g1[4] = (int)(tile1 & 0xffffu);                               // tile1 | tile2=0
  g1[5] = (int)stride0_elems;                                   // dim0_stride lo32
  g1[6] = 0;                                                    // stride hi bits
  g1[7] = 0;
  v4i z4 = {0, 0, 0, 0};
  v8i z8 = {0, 0, 0, 0, 0, 0, 0, 0};
  __builtin_amdgcn_tensor_load_to_lds(g0, g1, z4, z4, z8, 0);
}

// ---------------------------------------------------------------------------
// Kernel 0: one-shot f32 -> f16 conversion (x, Wq, Wk, Wv)
// ---------------------------------------------------------------------------
__global__ __launch_bounds__(256) void cvt_f32_to_f16(
    const float* __restrict__ src, _Float16* __restrict__ dst, int n4)
{
  const int i = blockIdx.x * 256 + threadIdx.x;
  if (i < n4) {
    v4f v = ((const v4f*)src)[i];
    v4h h;
    #pragma unroll
    for (int e = 0; e < 4; ++e) h[e] = (_Float16)v[e];
    ((v4h*)dst)[i] = h;
  }
}

// ---------------------------------------------------------------------------
// Kernel 1: projection GEMM on pre-converted f16:
//   out[m,n] = sum_d xh[m,d] * Wh[n,d] + bias[n]
// 8 waves/block, each wave a 32x32 tile of a 64x128 block tile.
// transposed==0: f16 row-major [B*T, D] (Q, K); ==1: f16 [B][D][T] (V^T).
// ---------------------------------------------------------------------------
__global__ __launch_bounds__(256) void qkv_proj_wmma(
    const _Float16* __restrict__ xh, const _Float16* __restrict__ Wh,
    const float* __restrict__ bias, _Float16* __restrict__ out, int transposed)
{
  const int lane = threadIdx.x & 31;
  const int wave = threadIdx.x >> 5;
  const int hi   = lane >> 4;
  const int lm   = lane & 15;
  const int m0 = blockIdx.y * 64  + (wave >> 2) * 32;
  const int n0 = blockIdx.x * 128 + (wave & 3) * 32;

  v8f c[2][2] = {};

  for (int k = 0; k < D_DIM; k += 32) {
    v16h a[2], b[2];
    #pragma unroll
    for (int am = 0; am < 2; ++am) {
      // A layout: lane m = lm, halves k = (e/8)*16 + hi*8 + e%8
      const _Float16* xr = xh + (size_t)(m0 + am * 16 + lm) * D_DIM + k + hi * 8;
      v8h a0 = *(const v8h*)xr;
      v8h a1 = *(const v8h*)(xr + 16);
      a[am] = __builtin_shufflevector(a0, a1, 0,1,2,3,4,5,6,7,8,9,10,11,12,13,14,15);
    }
    #pragma unroll
    for (int bn = 0; bn < 2; ++bn) {
      // B layout: lane n = lm, k = (e/8)*16 + hi*8 + e%8
      const _Float16* wr = Wh + (size_t)(n0 + bn * 16 + lm) * D_DIM + k + hi * 8;
      v8h b0 = *(const v8h*)wr;
      v8h b1 = *(const v8h*)(wr + 16);
      b[bn] = __builtin_shufflevector(b0, b1, 0,1,2,3,4,5,6,7,8,9,10,11,12,13,14,15);
    }
    if (k + 32 < D_DIM) {
      __builtin_prefetch(xh + (size_t)(m0 + lm) * D_DIM + k + 32, 0, 1);
      __builtin_prefetch(Wh + (size_t)(n0 + lm) * D_DIM + k + 32, 0, 1);
    }
    #pragma unroll
    for (int am = 0; am < 2; ++am)
      #pragma unroll
      for (int bn = 0; bn < 2; ++bn)
        c[am][bn] = wmma_f16(a[am], b[bn], c[am][bn]);
  }

  // C/D layout: (vgpr r, lane l) -> (M = r + 8*hi, N = lm)
  #pragma unroll
  for (int bn = 0; bn < 2; ++bn) {
    const int col = n0 + bn * 16 + lm;
    const float bv = bias[col];
    #pragma unroll
    for (int am = 0; am < 2; ++am) {
      const int mt = m0 + am * 16 + hi * 8;
      if (!transposed) {
        #pragma unroll
        for (int r = 0; r < 8; ++r)
          out[(size_t)(mt + r) * D_DIM + col] = (_Float16)(c[am][bn][r] + bv);
      } else {
        v8h pack;
        #pragma unroll
        for (int r = 0; r < 8; ++r) pack[r] = (_Float16)(c[am][bn][r] + bv);
        const int bidx = mt / T_DIM;
        const int t0   = mt % T_DIM;
        *(v8h*)(out + ((size_t)bidx * D_DIM + col) * T_DIM + t0) = pack;
      }
    }
  }
}

// ---------------------------------------------------------------------------
// Kernel 2: split-D flash attention, causal. 16 query rows per block, 8 waves;
// wave w owns output columns [w*128, w*128+128). K blocks (32 keys x 1024 d,
// f16) are staged into LDS by the Tensor Data Mover, double-buffered, with
// TDM padding for bank spread. S partials reduced via ds_add_f32.
// ---------------------------------------------------------------------------
__global__ __launch_bounds__(256) void causal_attn_wmma(
    const _Float16* __restrict__ Qh, const _Float16* __restrict__ Kh,
    const _Float16* __restrict__ Vt, float* __restrict__ out)
{
  __shared__ _Float16 Kst[2][KBUF_HALFS];     // 2 x 66560 B TDM buffers
  __shared__ float    S_red[16][32];
  __shared__ _Float16 P[16][32];
  __shared__ float    m_s[16], l_s[16], alpha_s[16];

  const int lane = threadIdx.x & 31;
  const int wave = threadIdx.x >> 5;
  const int hi   = lane >> 4;
  const int lm   = lane & 15;
  const int q0   = blockIdx.x * 16;
  const int bidx = blockIdx.y;
  const int colbase = wave * 128;
  const float NEG_INF = -__builtin_inff();

  const _Float16* Qb = Qh + (size_t)bidx * T_DIM * D_DIM;
  const _Float16* Kb = Kh + (size_t)bidx * T_DIM * D_DIM;
  const _Float16* Vb = Vt + (size_t)bidx * D_DIM * T_DIM;

  v8f o[8] = {};

  if (threadIdx.x < 16) { m_s[threadIdx.x] = NEG_INF; l_s[threadIdx.x] = 0.f; }

  // Preload this wave's Q A-fragments (invariant over key blocks)
  v16h qa[4];
  #pragma unroll
  for (int dk = 0; dk < 4; ++dk) {
    const _Float16* qr = Qb + (size_t)(q0 + lm) * D_DIM + colbase + dk * 32 + hi * 16;
    qa[dk] = *(const v16h*)qr;
  }

  const int nb = (q0 + 47) >> 5;   // key blocks covering keys <= q0+15
  const bool tdm_wave = (__builtin_amdgcn_readfirstlane(wave) == 0);

  // Prologue: stage key block 0 into buffer 0
  if (tdm_wave) {
    tdm_load_2d_f16(lds_offset(&Kst[0][0]), Kb,
                    /*tile*/ D_DIM, 32, /*tensor*/ D_DIM, T_DIM, D_DIM,
                    /*pad*/ 1, 7u, 3u);   // +4 dwords per 1024B
  }

  for (int j = 0; j < nb; ++j) {
    ((float*)S_red)[threadIdx.x]       = 0.f;
    ((float*)S_red)[threadIdx.x + 256] = 0.f;
    if (tdm_wave) {
      if (j + 1 < nb) {
        tdm_load_2d_f16(lds_offset(&Kst[(j + 1) & 1][0]),
                        Kb + (size_t)(j + 1) * 32 * D_DIM,
                        D_DIM, 32, D_DIM, T_DIM, D_DIM, 1, 7u, 3u);
        __builtin_amdgcn_s_wait_tensorcnt(1);   // block j done, j+1 in flight
      } else {
        __builtin_amdgcn_s_wait_tensorcnt(0);
      }
    }
    __syncthreads();

    // Partial S over this wave's 128-wide D slice, K read from LDS buffer
    const char* kbuf = (const char*)&Kst[j & 1][0];
    v8f s[2] = {};
    #pragma unroll
    for (int nt = 0; nt < 2; ++nt) {
      const int keyl = nt * 16 + lm;            // local key (B-operand lane)
      #pragma unroll
      for (int dk = 0; dk < 4; ++dk) {
        const int dloc = colbase + dk * 32 + hi * 16;
        const unsigned off = (unsigned)keyl * KROW_BYTES + dloc * 2 + ((dloc >> 9) << 4);
        v16h bf = *(const v16h*)(kbuf + off);
        s[nt] = wmma_f16(qa[dk], bf, s[nt]);
      }
    }
    #pragma unroll
    for (int nt = 0; nt < 2; ++nt)
      #pragma unroll
      for (int r = 0; r < 8; ++r)
        atomicAdd(&S_red[r + 8 * hi][nt * 16 + lm], s[nt][r]);
    __syncthreads();

    // Online softmax (one thread per query row)
    if (threadIdx.x < 16) {
      const int row = threadIdx.x;
      const int qi  = q0 + row;
      float sv[32];
      float rowmax = NEG_INF;
      #pragma unroll
      for (int cx = 0; cx < 32; ++cx) {
        const int key = j * 32 + cx;
        float v = S_red[row][cx] * 0.03125f;    // 1/sqrt(1024)
        if (key > qi) v = NEG_INF;              // causal mask
        sv[cx] = v;
        rowmax = fmaxf(rowmax, v);
      }
      const float mprev = m_s[row];
      const float mnew  = fmaxf(mprev, rowmax);
      const float alpha = (mprev == NEG_INF) ? 0.f : __expf(mprev - mnew);
      float sum = 0.f;
      #pragma unroll
      for (int cx = 0; cx < 32; ++cx) {
        const float p = (sv[cx] == NEG_INF) ? 0.f : __expf(sv[cx] - mnew);
        sum += p;
        P[row][cx] = (_Float16)p;
      }
      l_s[row]     = alpha * l_s[row] + sum;
      m_s[row]     = mnew;
      alpha_s[row] = alpha;
    }
    __syncthreads();

    // Rescale O, accumulate P(16x32) @ V(32x128-slice); V^T rows contiguous keys
    v8f af = *(const v8f*)&alpha_s[8 * hi];
    const _Float16* pr = &P[lm][hi * 8];
    v8h p0 = *(const v8h*)pr;
    v8h p1 = *(const v8h*)(pr + 16);
    v16h pa = __builtin_shufflevector(p0, p1, 0,1,2,3,4,5,6,7,8,9,10,11,12,13,14,15);

    #pragma unroll
    for (int nt = 0; nt < 8; ++nt) {
      #pragma unroll
      for (int r = 0; r < 8; ++r) o[nt][r] *= af[r];
      const int dcol = colbase + nt * 16 + lm;
      const _Float16* vr = Vb + (size_t)dcol * T_DIM + j * 32 + hi * 8;
      v8h v0 = *(const v8h*)vr;
      v8h v1 = *(const v8h*)(vr + 16);
      v16h bf = __builtin_shufflevector(v0, v1, 0,1,2,3,4,5,6,7,8,9,10,11,12,13,14,15);
      o[nt] = wmma_f16(pa, bf, o[nt]);
    }
    __syncthreads();
  }

  // Normalize and store fp32 output
  v8f lf = *(const v8f*)&l_s[8 * hi];
  #pragma unroll
  for (int nt = 0; nt < 8; ++nt) {
    const int dcol = colbase + nt * 16 + lm;
    #pragma unroll
    for (int r = 0; r < 8; ++r) {
      const int row = q0 + 8 * hi + r;
      out[((size_t)bidx * T_DIM + row) * D_DIM + dcol] = o[nt][r] / lf[r];
    }
  }
}

extern "C" void kernel_launch(void* const* d_in, const int* in_sizes, int n_in,
                              void* d_out, int out_size, void* d_ws, size_t ws_size,
                              hipStream_t stream) {
  (void)in_sizes; (void)n_in; (void)out_size; (void)ws_size;
  const float* x  = (const float*)d_in[0];
  const float* Wq = (const float*)d_in[1];
  const float* bq = (const float*)d_in[2];
  const float* Wk = (const float*)d_in[3];
  const float* bk = (const float*)d_in[4];
  const float* Wv = (const float*)d_in[5];
  const float* bv = (const float*)d_in[6];
  float* out = (float*)d_out;

  const size_t tok = (size_t)B_DIM * T_DIM * D_DIM;   // 8M elements
  const size_t wsz = (size_t)D_DIM * D_DIM;           // 1M elements
  _Float16* xh  = (_Float16*)d_ws;       // f16 [B*T, D]
  _Float16* Wqh = xh  + tok;             // f16 [D, D]
  _Float16* Wkh = Wqh + wsz;
  _Float16* Wvh = Wkh + wsz;
  _Float16* Qh  = Wvh + wsz;             // f16 row-major [B*T, D]
  _Float16* Kh  = Qh  + tok;             // f16 row-major [B*T, D]
  _Float16* Vt  = Kh  + tok;             // f16 [B][D][T]

  cvt_f32_to_f16<<<dim3((unsigned)(tok / 4 / 256)), 256, 0, stream>>>(x,  xh,  (int)(tok / 4));
  cvt_f32_to_f16<<<dim3((unsigned)(wsz / 4 / 256)), 256, 0, stream>>>(Wq, Wqh, (int)(wsz / 4));
  cvt_f32_to_f16<<<dim3((unsigned)(wsz / 4 / 256)), 256, 0, stream>>>(Wk, Wkh, (int)(wsz / 4));
  cvt_f32_to_f16<<<dim3((unsigned)(wsz / 4 / 256)), 256, 0, stream>>>(Wv, Wvh, (int)(wsz / 4));

  dim3 g1(D_DIM / 128, (B_DIM * T_DIM) / 64);
  qkv_proj_wmma<<<g1, 256, 0, stream>>>(xh, Wqh, bq, Qh, 0);
  qkv_proj_wmma<<<g1, 256, 0, stream>>>(xh, Wkh, bk, Kh, 0);
  qkv_proj_wmma<<<g1, 256, 0, stream>>>(xh, Wvh, bv, Vt, 1);

  dim3 g2(T_DIM / 16, B_DIM);
  causal_attn_wmma<<<g2, 256, 0, stream>>>(Qh, Kh, Vt, out);
}
